// NetBinary_21844203667761
// MI455X (gfx1250) — 8-bit matrix operand order repaired and checked by reading, compiled, not run on hardware
//
#include <hip/hip_runtime.h>

typedef __attribute__((ext_vector_type(8))) int v8i;
typedef __attribute__((ext_vector_type(4))) int v4i;
typedef __attribute__((ext_vector_type(2))) int v2i;

#define TPB 256


__global__ void fill_u32_kernel(unsigned int* p, unsigned int val, long long n) {
  long long i = (long long)blockIdx.x * blockDim.x + threadIdx.x;
  long long stride = (long long)gridDim.x * blockDim.x;
  for (; i < n; i += stride) p[i] = val;
}

__global__ void bn_stats_kernel(const float* __restrict__ x, int N, int C, int HW,
                                float* __restrict__ mean, float* __restrict__ inv) {
  int c = blockIdx.x;
  long long cnt = (long long)N * HW;
  float s = 0.f, s2 = 0.f;
  for (long long i = threadIdx.x; i < cnt; i += blockDim.x) {
    long long n = i / HW, r = i - n * (long long)HW;
    float v = x[((long long)n * C + c) * HW + r];
    s += v; s2 += v * v;
  }
  __shared__ float sh[TPB], sh2[TPB];
  sh[threadIdx.x] = s; sh2[threadIdx.x] = s2;
  __syncthreads();
  for (int st = TPB / 2; st > 0; st >>= 1) {
    if (threadIdx.x < st) { sh[threadIdx.x] += sh[threadIdx.x + st];
                            sh2[threadIdx.x] += sh2[threadIdx.x + st]; }
    __syncthreads();
  }
  if (threadIdx.x == 0) {
    float m = sh[0] / (float)cnt;
    float var = sh2[0] / (float)cnt - m * m;
    mean[c] = m;
    inv[c]  = rsqrtf(var + 1e-5f);
  }
}

__global__ void bn_sign_abs_kernel(const float* __restrict__ x,
                                   const float* __restrict__ mean,
                                   const float* __restrict__ inv,
                                   const float* __restrict__ g,
                                   const float* __restrict__ b,
                                   signed char* __restrict__ s,
                                   float* __restrict__ abssum,
                                   int C, int HW, long long total) {
  long long i = (long long)blockIdx.x * blockDim.x + threadIdx.x;
  long long stride = (long long)gridDim.x * blockDim.x;
  float acc = 0.f;
  for (; i < total; i += stride) {
    int c = (int)((i / HW) % C);
    float xn = (x[i] - mean[c]) * inv[c] * g[c] + b[c];
    s[i] = (signed char)(xn > 0.f ? 1 : (xn < 0.f ? -1 : 0));
    acc += fabsf(xn);
  }
  __shared__ float sh[TPB];
  sh[threadIdx.x] = acc;
  __syncthreads();
  for (int st = TPB / 2; st > 0; st >>= 1) {
    if (threadIdx.x < st) sh[threadIdx.x] += sh[threadIdx.x + st];
    __syncthreads();
  }
  if (threadIdx.x == 0) atomicAdd(abssum, sh[0]);
}

__global__ void sign_abs_kernel(const float* __restrict__ x,
                                signed char* __restrict__ s,
                                float* __restrict__ abssum, long long total) {
  long long i = (long long)blockIdx.x * blockDim.x + threadIdx.x;
  long long stride = (long long)gridDim.x * blockDim.x;
  float acc = 0.f;
  for (; i < total; i += stride) {
    float v = x[i];
    s[i] = (signed char)(v > 0.f ? 1 : (v < 0.f ? -1 : 0));
    acc += fabsf(v);
  }
  __shared__ float sh[TPB];
  sh[threadIdx.x] = acc;
  __syncthreads();
  for (int st = TPB / 2; st > 0; st >>= 1) {
    if (threadIdx.x < st) sh[threadIdx.x] += sh[threadIdx.x + st];
    __syncthreads();
  }
  if (threadIdx.x == 0) atomicAdd(abssum, sh[0]);
}

__global__ void pack_w_kernel(const float* __restrict__ w, signed char* __restrict__ wp,
                              int OC, int Kreal, int Kpad, int OCpad) {
  long long total = (long long)OCpad * Kpad;
  long long i = (long long)blockIdx.x * blockDim.x + threadIdx.x;
  long long stride = (long long)gridDim.x * blockDim.x;
  for (; i < total; i += stride) {
    int oc = (int)(i / Kpad), k = (int)(i - (long long)oc * Kpad);
    float v = (oc < OC && k < Kreal) ? w[(long long)oc * Kreal + k] : 0.f;
    wp[i] = (signed char)(v > 0.f ? 1 : (v < 0.f ? -1 : 0));
  }
}

__global__ void im2col_sign_kernel(const signed char* __restrict__ s,
                                   signed char* __restrict__ A,
                                   int Cin, int IH, int IW, int OH, int OW,
                                   int Kreal, int Kpad, long long total) {
  long long i = (long long)blockIdx.x * blockDim.x + threadIdx.x;
  long long stride = (long long)gridDim.x * blockDim.x;
  for (; i < total; i += stride) {
    int m = (int)(i / Kpad);
    int k = (int)(i - (long long)m * Kpad);
    signed char v = 0;
    if (k < Kreal) {
      int c  = k / 9;
      int r9 = k - c * 9;
      int kh = r9 / 3, kw = r9 - kh * 3;
      int img = m / (OH * OW);
      int r   = m - img * (OH * OW);
      int oh = r / OW, ow = r - oh * OW;
      v = s[(((long long)img * Cin + c) * IH + (oh + kh)) * IW + (ow + kw)];
    }
    A[i] = v;
  }
}

__global__ void prelu_kernel(float* __restrict__ x, const float* __restrict__ a,
                             long long n) {
  long long i = (long long)blockIdx.x * blockDim.x + threadIdx.x;
  long long stride = (long long)gridDim.x * blockDim.x;
  float alpha = a[0];
  for (; i < n; i += stride) {
    float v = x[i];
    x[i] = v >= 0.f ? v : alpha * v;
  }
}

__device__ __forceinline__ void atomicMaxF32(float* addr, float v) {
  if (v >= 0.f) atomicMax((int*)addr, __float_as_int(v));
  else          atomicMin((unsigned int*)addr, __float_as_uint(v));
}

template <int NT>
__device__ __forceinline__ void bin_mma_xN(const signed char* __restrict__ aBase,
                                           const signed char* __restrict__ bBase,
                                           long long bStride, int K, v8i acc[NT]) {
  for (int k0 = 0; k0 < K; k0 += 64) {
    v4i a03 = *(const v4i*)(aBase + k0);
    v4i a47 = *(const v4i*)(aBase + k0 + 32);
    v8i bfrag[NT];
#pragma unroll
    for (int t = 0; t < NT; ++t) {
      const signed char* bp = bBase + (long long)t * bStride;
      v4i b03 = *(const v4i*)(bp + k0);
      v4i b47 = *(const v4i*)(bp + k0 + 32);
      v8i b = {b03[0], b03[1], b03[2], b03[3], b47[0], b47[1], b47[2], b47[3]};
      bfrag[t] = b;
    }
    v8i a = {a03[0], a03[1], a03[2], a03[3], a47[0], a47[1], a47[2], a47[3]};
#pragma unroll
    for (int t = 0; t < NT; ++t) {
      acc[t] = __builtin_amdgcn_wmma_i32_16x16x64_iu8(true, a, true, bfrag[t],
                                                      acc[t], false, false);
    }
  }
}

template <int NT>
__global__ __launch_bounds__(256) void gemm_conv_kernel(
    const signed char* __restrict__ A,
    const signed char* __restrict__ wp,
    float* __restrict__ pooled,
    const float* __restrict__ abssum, float invCount,
    int OC, int OH, int OW, int PH, int PW, int Kpad, int Mreal) {
  const int lane = threadIdx.x & 31;
  const int wv   = threadIdx.x >> 5;
  const int Mtiles = Mreal >> 4;
  const int tileM  = blockIdx.x * (blockDim.x >> 5) + wv;
  const int tileN0 = blockIdx.y * NT;
  if (tileM >= Mtiles) return;

  const int rowA = tileM * 16 + (lane & 15);
  const signed char* aBase = A  + (long long)rowA * Kpad + ((lane >> 4) << 4);
  const signed char* bBase = wp + (long long)(tileN0 * 16 + (lane & 15)) * Kpad
                                + ((lane >> 4) << 4);
  const long long bStride = 16ll * Kpad;

  v8i acc[NT];
  const v8i z = {0, 0, 0, 0, 0, 0, 0, 0};
#pragma unroll
  for (int t = 0; t < NT; ++t) acc[t] = z;

  bin_mma_xN<NT>(aBase, bBase, bStride, Kpad, acc);

  const float scale = abssum[0] * invCount;
#pragma unroll
  for (int r = 0; r < 8; ++r) {
    int m  = tileM * 16 + r + ((lane >> 4) << 3);
    int im = m / (OH * OW);
    int rr = m - im * (OH * OW);
    int oh = rr / OW, ow = rr - oh * OW;
    int ph = oh >> 1, pw = ow >> 1;
    if (ph >= PH || pw >= PW) continue;
    long long base = (((long long)im * OC) * PH + ph) * PW + pw;
#pragma unroll
    for (int t = 0; t < NT; ++t) {
      int oc = (tileN0 + t) * 16 + (lane & 15);
      atomicMaxF32(&pooled[base + (long long)oc * PH * PW],
                   (float)acc[t][r] * scale);
    }
  }
}

template <int NT>
__global__ __launch_bounds__(256) void gemm_fc_kernel(
    const signed char* __restrict__ sa,
    const signed char* __restrict__ wp,
    float* __restrict__ out,
    const float* __restrict__ abssum, float invCount,
    int M, int K, int Nreal) {
  const int lane = threadIdx.x & 31;
  const int wv   = threadIdx.x >> 5;
  const int Mtiles = M >> 4;
  const int tileM  = blockIdx.x * (blockDim.x >> 5) + wv;
  const int tileN0 = blockIdx.y * NT;
  if (tileM >= Mtiles) return;

  const int rowA = tileM * 16 + (lane & 15);
  const signed char* aBase = sa + (long long)rowA * K + ((lane >> 4) << 4);
  const signed char* bBase = wp + (long long)(tileN0 * 16 + (lane & 15)) * K
                                + ((lane >> 4) << 4);
  const long long bStride = 16ll * K;

  v8i acc[NT];
  const v8i z = {0, 0, 0, 0, 0, 0, 0, 0};
#pragma unroll
  for (int t = 0; t < NT; ++t) acc[t] = z;

  bin_mma_xN<NT>(aBase, bBase, bStride, K, acc);

  const float scale = abssum[0] * invCount;
#pragma unroll
  for (int t = 0; t < NT; ++t) {
    int n = (tileN0 + t) * 16 + (lane & 15);
    if (n < Nreal) {
#pragma unroll
      for (int r = 0; r < 8; ++r) {
        int m = tileM * 16 + r + ((lane >> 4) << 3);
        out[(long long)m * Nreal + n] = (float)acc[t][r] * scale;
      }
    }
  }
}


static inline dim3 ew_grid(long long n) {
  long long b = (n + TPB - 1) / TPB;
  if (b > 32768) b = 32768;
  return dim3((unsigned)b);
}

extern "C" void kernel_launch(void* const* d_in, const int* in_sizes, int n_in,
                              void* d_out, int out_size, void* d_ws, size_t ws_size,
                              hipStream_t stream) {
  (void)in_sizes; (void)n_in; (void)out_size; (void)ws_size;

  const float* x   = (const float*)d_in[0];
  const float* g0  = (const float*)d_in[1];
  const float* b0  = (const float*)d_in[2];
  const float* w0  = (const float*)d_in[3];
  const float* a0  = (const float*)d_in[4];
  const float* g1  = (const float*)d_in[5];
  const float* b1  = (const float*)d_in[6];
  const float* w1  = (const float*)d_in[7];
  const float* a1  = (const float*)d_in[8];
  const float* g2  = (const float*)d_in[9];
  const float* b2  = (const float*)d_in[10];
  const float* w2  = (const float*)d_in[11];
  const float* a2  = (const float*)d_in[12];
  const float* fw0 = (const float*)d_in[13];
  const float* fw1 = (const float*)d_in[14];
  const float* fw2 = (const float*)d_in[15];
  float* out = (float*)d_out;

  char* ws = (char*)d_ws;
  float*       absS  = (float*)ws;
  float*       meanB = (float*)(ws + 1024);
  float*       invB  = (float*)(ws + 4096);
  signed char* signB = (signed char*)(ws + (1ll << 16));
  signed char* wpack = (signed char*)(ws + (1ll << 16) + (8ll << 20));
  signed char* convA = wpack + (2ll << 20);
  float*       P0    = (float*)(ws + (1ll << 16) + (84ll << 20));
  float*       P1    = (float*)(ws + (1ll << 16) + (116ll << 20));

  {
    const long long total = 64ll * 3 * 64 * 64;
    const long long pool  = 64ll * 128 * 31 * 31;
    const long long aelem = 246016ll * 64;
    fill_u32_kernel<<<1, TPB, 0, stream>>>((unsigned*)absS, 0u, 1);
    bn_stats_kernel<<<3, TPB, 0, stream>>>(x, 64, 3, 64 * 64, meanB, invB);
    bn_sign_abs_kernel<<<ew_grid(total), TPB, 0, stream>>>(x, meanB, invB, g0, b0,
                                                           signB, absS, 3, 64 * 64, total);
    im2col_sign_kernel<<<ew_grid(aelem), TPB, 0, stream>>>(signB, convA,
                                                           3, 64, 64, 62, 62, 27, 64, aelem);
    pack_w_kernel<<<ew_grid(128ll * 64), TPB, 0, stream>>>(w0, wpack, 128, 27, 64, 128);
    fill_u32_kernel<<<ew_grid(pool), TPB, 0, stream>>>((unsigned*)P0, 0xFF800000u, pool);
    gemm_conv_kernel<4><<<dim3(15376 / 8, 2), 256, 0, stream>>>(
        convA, wpack, P0, absS, 1.f / (float)total,
        128, 62, 62, 31, 31, 64, 246016);
    prelu_kernel<<<ew_grid(pool), TPB, 0, stream>>>(P0, a0, pool);
  }

  {
    const long long total = 64ll * 128 * 31 * 31;
    const long long pool  = 64ll * 256 * 14 * 14;
    const long long aelem = 53824ll * 1152;
    fill_u32_kernel<<<1, TPB, 0, stream>>>((unsigned*)absS, 0u, 1);
    bn_stats_kernel<<<128, TPB, 0, stream>>>(P0, 64, 128, 31 * 31, meanB, invB);
    bn_sign_abs_kernel<<<ew_grid(total), TPB, 0, stream>>>(P0, meanB, invB, g1, b1,
                                                           signB, absS, 128, 31 * 31, total);
    im2col_sign_kernel<<<ew_grid(aelem), TPB, 0, stream>>>(signB, convA,
                                                           128, 31, 31, 29, 29, 1152, 1152, aelem);
    pack_w_kernel<<<ew_grid(256ll * 1152), TPB, 0, stream>>>(w1, wpack, 256, 1152, 1152, 256);
    fill_u32_kernel<<<ew_grid(pool), TPB, 0, stream>>>((unsigned*)P1, 0xFF800000u, pool);
    gemm_conv_kernel<4><<<dim3((3364 + 7) / 8, 4), 256, 0, stream>>>(
        convA, wpack, P1, absS, 1.f / (float)total,
        256, 29, 29, 14, 14, 1152, 53824);
    prelu_kernel<<<ew_grid(pool), TPB, 0, stream>>>(P1, a1, pool);
  }

  {
    const long long total = 64ll * 256 * 14 * 14;
    const long long pool  = 64ll * 512 * 6 * 6;
    const long long aelem = 9216ll * 2304;
    fill_u32_kernel<<<1, TPB, 0, stream>>>((unsigned*)absS, 0u, 1);
    bn_stats_kernel<<<256, TPB, 0, stream>>>(P1, 64, 256, 14 * 14, meanB, invB);
    bn_sign_abs_kernel<<<ew_grid(total), TPB, 0, stream>>>(P1, meanB, invB, g2, b2,
                                                           signB, absS, 256, 14 * 14, total);
    im2col_sign_kernel<<<ew_grid(aelem), TPB, 0, stream>>>(signB, convA,
                                                           256, 14, 14, 12, 12, 2304, 2304, aelem);
    pack_w_kernel<<<ew_grid(512ll * 2304), TPB, 0, stream>>>(w2, wpack, 512, 2304, 2304, 512);
    fill_u32_kernel<<<ew_grid(pool), TPB, 0, stream>>>((unsigned*)P0, 0xFF800000u, pool);
    gemm_conv_kernel<4><<<dim3(576 / 8, 8), 256, 0, stream>>>(
        convA, wpack, P0, absS, 1.f / (float)total,
        512, 12, 12, 6, 6, 2304, 9216);
    prelu_kernel<<<ew_grid(pool), TPB, 0, stream>>>(P0, a2, pool);
  }

  {
    const long long total = 64ll * 18432;
    fill_u32_kernel<<<1, TPB, 0, stream>>>((unsigned*)absS, 0u, 1);
    sign_abs_kernel<<<ew_grid(total), TPB, 0, stream>>>(P0, signB, absS, total);
    pack_w_kernel<<<ew_grid(4096ll * 18432), TPB, 0, stream>>>(fw0, wpack, 4096, 18432, 18432, 4096);
    gemm_fc_kernel<4><<<dim3(1, 64), 128, 0, stream>>>(signB, wpack, P1, absS,
                                                       1.f / (float)total, 64, 18432, 4096);
  }

  {
    const long long total = 64ll * 4096;
    fill_u32_kernel<<<1, TPB, 0, stream>>>((unsigned*)absS, 0u, 1);
    sign_abs_kernel<<<ew_grid(total), TPB, 0, stream>>>(P1, signB, absS, total);
    pack_w_kernel<<<ew_grid(1024ll * 4096), TPB, 0, stream>>>(fw1, wpack, 1024, 4096, 4096, 1024);
    gemm_fc_kernel<4><<<dim3(1, 16), 128, 0, stream>>>(signB, wpack, P0, absS,
                                                       1.f / (float)total, 64, 4096, 1024);
  }

  {
    const long long total = 64ll * 1024;
    fill_u32_kernel<<<1, TPB, 0, stream>>>((unsigned*)absS, 0u, 1);
    sign_abs_kernel<<<ew_grid(total), TPB, 0, stream>>>(P0, signB, absS, total);
    pack_w_kernel<<<ew_grid(16ll * 1024), TPB, 0, stream>>>(fw2, wpack, 10, 1024, 1024, 16);
    gemm_fc_kernel<1><<<dim3(1, 1), 128, 0, stream>>>(signB, wpack, out, absS,
                                                      1.f / (float)total, 64, 1024, 10);
  }
}
